// GCNRegressor_9869834846567
// MI455X (gfx1250) — matrix operand order repaired and checked by reading, not compiled, not run on hardware
//
#include <hip/hip_runtime.h>

#define NNODES 100000
#define NEDGES 1600000
#define NGRAPH 512
#define FIN    30
#define HDIM   128
#define BN_EPS 1e-5f

typedef __attribute__((ext_vector_type(4)))  _Float16 v4h;
typedef __attribute__((ext_vector_type(8)))  _Float16 v8h;
typedef __attribute__((ext_vector_type(16))) _Float16 v16h;
typedef __attribute__((ext_vector_type(8)))  float    v8f;

// ---------------- degree / dinv ----------------
__global__ void k_deg_init(float* __restrict__ deg) {
  int i = blockIdx.x * blockDim.x + threadIdx.x;
  if (i < NNODES) deg[i] = 1.0f;                 // self-loop contributes 1
}
__global__ void k_deg_accum(const int* __restrict__ dst, float* __restrict__ deg) {
  int e = blockIdx.x * blockDim.x + threadIdx.x;
  if (e < NEDGES) atomicAdd(&deg[dst[e]], 1.0f);
}
__global__ void k_dinv(float* __restrict__ deg) {
  int i = blockIdx.x * blockDim.x + threadIdx.x;
  if (i < NNODES) deg[i] = rsqrtf(fmaxf(deg[i], 1.0f));
}

// ---------------- operand packing (f32 -> f16, K-padded / transposed) ----------------
// Layer-0 input: x[N x 30] f32 -> xh[N x 32] f16 (K padded with zeros)
__global__ void k_pack_x0(const float* __restrict__ x, _Float16* __restrict__ xh) {
  int idx = blockIdx.x * blockDim.x + threadIdx.x;
  if (idx >= NNODES * 32) return;
  int r = idx >> 5, k = idx & 31;
  xh[idx] = (k < FIN) ? (_Float16)x[r * FIN + k] : (_Float16)0.0f;
}
// Weights: W[din x 128] f32 -> Wt[128 x kpad] f16 (transposed so B loads are contiguous in K)
__global__ void k_pack_w(const float* __restrict__ W, int din, int kpad,
                         _Float16* __restrict__ Wt) {
  int idx = blockIdx.x * blockDim.x + threadIdx.x;
  if (idx >= HDIM * kpad) return;
  int n = idx / kpad, k = idx - n * kpad;
  Wt[idx] = (k < din) ? (_Float16)W[k * HDIM + n] : (_Float16)0.0f;
}

// ---------------- WMMA GEMM: Hout[N x 128](f16) = Xh[N x KPAD] * Wt^T ----------------
// 8 waves per block share one 16-row tile; wave w computes the 16x16 tile at column 16*w.
// All fragment loads are contiguous vector loads; no predication -> EXEC all-ones (WMMA req).
template<int KPAD>
__global__ __launch_bounds__(256) void k_gemm_wmma(const _Float16* __restrict__ Xh,
                                                   const _Float16* __restrict__ Wt,
                                                   _Float16* __restrict__ Hout) {
  const int tile_m = blockIdx.x;            // 0..6249 (N = 6250*16 exactly)
  const int wave   = threadIdx.x >> 5;      // 0..7 = column tile
  const int lane   = threadIdx.x & 31;
  const int g      = lane >> 4;             // half-wave group
  const int mn     = lane & 15;
  const int m      = tile_m * 16 + mn;      // A row for this lane
  const int n      = wave * 16 + mn;        // B col for this lane

  const _Float16* xrow = Xh + (size_t)m * KPAD;
  const _Float16* wrow = Wt + (size_t)n * KPAD;

  v8f c = {};
#pragma unroll
  for (int k0 = 0; k0 < KPAD; k0 += 32) {
    // A fragment: halves i -> K = k0 + g*16 + i (contiguous, matches B)
    v8h alo = *(const v8h*)(xrow + k0 + g * 16);
    v8h ahi = *(const v8h*)(xrow + k0 + g * 16 + 8);
    v16h a = __builtin_shufflevector(alo, ahi, 0, 1, 2, 3, 4, 5, 6, 7,
                                               8, 9, 10, 11, 12, 13, 14, 15);
    // B fragment: lane column n, halves i -> K = k0 + g*16 + i (contiguous in Wt)
    v16h b = *(const v16h*)(wrow + k0 + g * 16);
    c = __builtin_amdgcn_wmma_f32_16x16x32_f16(false, a, false, b, (short)0, c, false, false);
  }
  // C/D layout: VGPR v holds row M = v + 8*g, col = lane&15; store h as f16 (halves gather BW)
#pragma unroll
  for (int v = 0; v < 8; ++v)
    Hout[(size_t)(tile_m * 16 + v + 8 * g) * HDIM + wave * 16 + mn] = (_Float16)c[v];
}

// ---------------- agg = bias + self-loop term (writes every element) ----------------
__global__ void k_agg_init(const _Float16* __restrict__ h, const float* __restrict__ dinv,
                           const float* __restrict__ bias, float* __restrict__ agg) {
  int idx = blockIdx.x * blockDim.x + threadIdx.x;
  if (idx >= NNODES * HDIM) return;
  int r = idx >> 7, ch = idx & 127;
  float di = dinv[r];
  agg[idx] = bias[ch] + (float)h[idx] * di * di;
}

// ---------------- edge scatter: one wave per edge, 4 f16 per lane (128 cols) ----------------
__global__ void k_edge_scatter(const _Float16* __restrict__ h,
                               const int* __restrict__ src, const int* __restrict__ dst,
                               const float* __restrict__ dinv, float* __restrict__ agg) {
  int e = blockIdx.x * (blockDim.x >> 5) + (threadIdx.x >> 5);
  if (e >= NEDGES) return;
  int lane = threadIdx.x & 31;
  int s = src[e], d = dst[e];
  float nrm = dinv[s] * dinv[d];
  v4h v = ((const v4h*)(h + (size_t)s * HDIM))[lane];      // b64 load, coalesced per wave
  float* ad = agg + (size_t)d * HDIM + lane * 4;
  atomicAdd(ad + 0, (float)v[0] * nrm);
  atomicAdd(ad + 1, (float)v[1] * nrm);
  atomicAdd(ad + 2, (float)v[2] * nrm);
  atomicAdd(ad + 3, (float)v[3] * nrm);
}

// ---------------- BN stats (biased) + apply + ReLU (writes f16 for next GEMM) ----------------
__global__ void k_zero(float* __restrict__ p, int n) {
  int i = blockIdx.x * blockDim.x + threadIdx.x;
  if (i < n) p[i] = 0.0f;
}
__global__ void k_bn_stats(const float* __restrict__ agg, float* __restrict__ sums,
                           float* __restrict__ sumsq) {
  int c = threadIdx.x;                       // 128 channels, coalesced row reads
  float s = 0.0f, sq = 0.0f;
  for (int r = blockIdx.x; r < NNODES; r += gridDim.x) {
    float v = agg[r * HDIM + c];
    s += v; sq += v * v;
  }
  atomicAdd(&sums[c], s);
  atomicAdd(&sumsq[c], sq);
}
__global__ void k_bn_apply(const float* __restrict__ agg, const float* __restrict__ sums,
                           const float* __restrict__ sumsq, const float* __restrict__ gamma,
                           const float* __restrict__ beta, _Float16* __restrict__ xout) {
  int idx = blockIdx.x * blockDim.x + threadIdx.x;
  if (idx >= NNODES * HDIM) return;
  int c = idx & 127;
  const float inv_n = 1.0f / (float)NNODES;
  float mu  = sums[c] * inv_n;
  float var = sumsq[c] * inv_n - mu * mu;
  float y = (agg[idx] - mu) * rsqrtf(var + BN_EPS) * gamma[c] + beta[c];
  xout[idx] = (_Float16)fmaxf(y, 0.0f);
}

// ---------------- global mean pool (from f16 activations) ----------------
__global__ void k_pool_accum(const _Float16* __restrict__ x, const int* __restrict__ batch,
                             float* __restrict__ pooled, float* __restrict__ counts) {
  int c = threadIdx.x;                       // 128 channels
  for (int r = blockIdx.x; r < NNODES; r += gridDim.x) {
    int gph = batch[r];
    atomicAdd(&pooled[gph * HDIM + c], (float)x[r * HDIM + c]);
    if (c == 0) atomicAdd(&counts[gph], 1.0f);
  }
}

// ---------------- MLP head: 128 -> 64 -> relu -> 1 ----------------
__global__ __launch_bounds__(64) void k_mlp(const float* __restrict__ pooled,
                                            const float* __restrict__ counts,
                                            const float* __restrict__ HW1, const float* __restrict__ Hb1,
                                            const float* __restrict__ HW2, const float* __restrict__ Hb2,
                                            float* __restrict__ out) {
  int gph = blockIdx.x;
  int j = threadIdx.x;                        // 0..63
  __shared__ float p[HDIM];
  __shared__ float red[64];
  float cnt = fmaxf(counts[gph], 1.0f);
  for (int c = j; c < HDIM; c += 64) p[c] = pooled[gph * HDIM + c] / cnt;
  __syncthreads();
  float acc = Hb1[j];
#pragma unroll 8
  for (int k = 0; k < HDIM; ++k) acc += p[k] * HW1[k * 64 + j];
  acc = fmaxf(acc, 0.0f);
  red[j] = acc * HW2[j];
  __syncthreads();
  for (int s = 32; s > 0; s >>= 1) {
    if (j < s) red[j] += red[j + s];
    __syncthreads();
  }
  if (j == 0) out[gph] = red[0] + Hb2[0];
}

// ---------------- host side ----------------
extern "C" void kernel_launch(void* const* d_in, const int* in_sizes, int n_in,
                              void* d_out, int out_size, void* d_ws, size_t ws_size,
                              hipStream_t stream) {
  (void)in_sizes; (void)n_in; (void)out_size; (void)ws_size;

  const float* x_in  = (const float*)d_in[0];            // [N, 30]
  const int*   ei    = (const int*)d_in[1];              // [2, E]
  const int*   batch = (const int*)d_in[2];              // [N]
  const float* Wl[3]  = { (const float*)d_in[3],  (const float*)d_in[7],  (const float*)d_in[11] };
  const float* bl[3]  = { (const float*)d_in[4],  (const float*)d_in[8],  (const float*)d_in[12] };
  const float* gl[3]  = { (const float*)d_in[5],  (const float*)d_in[9],  (const float*)d_in[13] };
  const float* btl[3] = { (const float*)d_in[6],  (const float*)d_in[10], (const float*)d_in[14] };
  const float* HW1 = (const float*)d_in[15];
  const float* Hb1 = (const float*)d_in[16];
  const float* HW2 = (const float*)d_in[17];
  const float* Hb2 = (const float*)d_in[18];
  float* out = (float*)d_out;

  // workspace carve-up (256B aligned); total ~110 MB
  char* ws = (char*)d_ws;
  size_t off = 0;
  auto carve = [&](size_t bytes) -> char* {
    char* p = ws + off;
    off += (bytes + 255) & ~(size_t)255;
    return p;
  };
  float*    dinv   = (float*)carve((size_t)NNODES * 4);
  float*    stats  = (float*)carve(256 * 4);             // sums[128] | sumsq[128]
  float*    pooled = (float*)carve((size_t)(NGRAPH * HDIM + NGRAPH) * 4);
  _Float16* x0h    = (_Float16*)carve((size_t)NNODES * 32 * 2);          // layer-0 A, K padded to 32
  _Float16* xh     = (_Float16*)carve((size_t)NNODES * HDIM * 2);        // layer 1/2 A (BN output)
  _Float16* wt     = (_Float16*)carve((size_t)HDIM * HDIM * 2);          // packed/transposed weights
  _Float16* bufH   = (_Float16*)carve((size_t)NNODES * HDIM * 2);        // GEMM output h (f16)
  float*    bufA   = (float*)carve((size_t)NNODES * HDIM * 4);           // aggregation (f32 atomics)
  float*    sums   = stats;
  float*    sumsq  = stats + 128;
  float*    counts = pooled + NGRAPH * HDIM;

  const int* src = ei;              // row 0
  const int* dst = ei + NEDGES;     // row 1

  // degrees -> dinv
  k_deg_init<<<(NNODES + 255) / 256, 256, 0, stream>>>(dinv);
  k_deg_accum<<<(NEDGES + 255) / 256, 256, 0, stream>>>(dst, dinv);
  k_dinv<<<(NNODES + 255) / 256, 256, 0, stream>>>(dinv);

  // pack layer-0 input once
  k_pack_x0<<<(NNODES * 32 + 255) / 256, 256, 0, stream>>>(x_in, x0h);

  const int elem_blocks = (NNODES * HDIM + 255) / 256;

  for (int l = 0; l < 3; ++l) {
    int din  = (l == 0) ? FIN : HDIM;
    int kpad = (l == 0) ? 32  : HDIM;
    k_pack_w<<<(HDIM * kpad + 255) / 256, 256, 0, stream>>>(Wl[l], din, kpad, wt);
    if (l == 0)
      k_gemm_wmma<32><<<NNODES / 16, 256, 0, stream>>>(x0h, wt, bufH);
    else
      k_gemm_wmma<HDIM><<<NNODES / 16, 256, 0, stream>>>(xh, wt, bufH);
    k_agg_init<<<elem_blocks, 256, 0, stream>>>(bufH, dinv, bl[l], bufA);
    k_edge_scatter<<<(NEDGES + 7) / 8, 256, 0, stream>>>(bufH, src, dst, dinv, bufA);
    k_zero<<<1, 256, 0, stream>>>(stats, 256);
    k_bn_stats<<<512, 128, 0, stream>>>(bufA, sums, sumsq);
    k_bn_apply<<<elem_blocks, 256, 0, stream>>>(bufA, sums, sumsq, gl[l], btl[l], xh);
  }

  k_zero<<<(NGRAPH * HDIM + NGRAPH + 255) / 256, 256, 0, stream>>>(pooled, NGRAPH * HDIM + NGRAPH);
  k_pool_accum<<<512, 128, 0, stream>>>(xh, batch, pooled, counts);
  k_mlp<<<NGRAPH, 64, 0, stream>>>(pooled, counts, HW1, Hb1, HW2, Hb2, out);
}
